// KrauseAttention_49143015801132
// MI455X (gfx1250) — matrix operand order repaired and checked by reading, not compiled, not run on hardware
//
#include <hip/hip_runtime.h>
#include <math.h>

// ---------------------------------------------------------------------------
// KrauseAttention for MI455X (gfx1250, wave32, WMMA + TDM).
//   B=4, T=2048, E=512, H=8, D=64.  IMAGE_C=1 => causal mask is plain k<=q.
//   WINDOW=128 => each query sees keys [q-127, q]; TOP_K=96 of those.
//
// Pipeline:
//   0) one-time f32->bf16 conversion of x and Wq/Wk/Wv/Wo (hardware v_cvt)
//   1) QKV projections: bf16 WMMA GEMM; weight K-slabs (512x32) staged into
//      LDS by the Tensor Data Mover (tensor_load_to_lds, TENSORcnt),
//      double-buffered so DMA overlaps WMMA.
//   2) windowed Gaussian attention, per-row top-96 via 32-bit radix select,
//      softmax, P.V via WMMA
//   3) output projection: same TDM-staged WMMA GEMM -> f32
// ---------------------------------------------------------------------------

typedef __attribute__((ext_vector_type(16))) __bf16 v16bf;
typedef __attribute__((ext_vector_type(8)))  float  v8f;
typedef __attribute__((ext_vector_type(4)))  unsigned int v4u;
typedef __attribute__((ext_vector_type(8)))  int v8i;
typedef __attribute__((ext_vector_type(4)))  int v4i;

#define B_    4
#define T_    2048
#define E_    512
#define H_    8
#define D_    64
#define M_    (B_*T_)          // 8192 rows
#define WIN_  128
#define TOPK_ 96
#define MAXKT 9                // max 16-key tiles covering a 16-query window
#define PCOLS 160              // MAXKT*16 padded to a 32-key chunk multiple
#define NCH   (E_/32)          // 16 K-chunks per GEMM

#if __has_builtin(__builtin_amdgcn_tensor_load_to_lds) && \
    __has_builtin(__builtin_amdgcn_s_wait_tensorcnt)
#define HAVE_TDM 1
#endif

// The compiler does not model tensor_load_to_lds as an LDS writer.  A bare
// "memory"-clobber asm is NOT enough: the LDS global's address never escapes,
// so globals-AA still folds its (never-stored) loads to undef.  Passing the
// LDS base pointer INTO the asm escapes it, making the asm a legal potential
// writer through that pointer -- the subsequent ds_loads must stay.
static __device__ __forceinline__ void tdm_lds_publish(const void* lds_base) {
  asm volatile("" : : "v"(lds_base) : "memory");
}

// monotonic float<->uint mapping for radix select (bit-exact ordering)
static __device__ __forceinline__ unsigned sortkey(float f) {
  union { float f; unsigned u; } a; a.f = f;
  return (a.u & 0x80000000u) ? ~a.u : (a.u | 0x80000000u);
}
static __device__ __forceinline__ float keyfloat(unsigned k) {
  union { unsigned u; float f; } a;
  a.u = (k & 0x80000000u) ? (k & 0x7FFFFFFFu) : ~k;
  return a.f;
}

#if defined(HAVE_TDM)
// ---------------------------------------------------------------------------
// Issue a TDM load of W[0:512, kb:kb+32] (bf16, row stride E_) into LDS at
// byte offset lds_off.  D# packing per CDNA5 ISA ch.8 (group0/group1):
//   g0: count=1 | lds_addr | global_addr | type=2
//   g1: data_size=2B, tensor_dim0=512, tensor_dim1=512, tile=32x512,
//       tensor_dim0_stride=512
// 2D tensor => groups 2/3 unused (zero-filled).
// ---------------------------------------------------------------------------
static __device__ __forceinline__ void
tdm_issue_w_slab(const __bf16* Wg, int kb, unsigned lds_off) {
  unsigned long long ga = (unsigned long long)Wg + (unsigned long long)(kb * 2);
  v4u g0;
  g0[0] = 1u;                                            // count=1, load, user
  g0[1] = lds_off;                                       // lds_addr (bytes)
  g0[2] = (unsigned)ga;                                  // global_addr[31:0]
  g0[3] = ((unsigned)(ga >> 32) & 0x01FFFFFFu) | 0x80000000u; // addr[56:32]|type=2
  v8i g1;
  g1[0] = 0x00010000;                                    // data_size=1 (2 bytes)
  g1[1] = (int)(512u << 16);                             // tensor_dim0 = 512
  g1[2] = (int)(512u << 16);                             // tensor_dim1 = 512
  g1[3] = (int)(32u  << 16);                             // tile_dim0 = 32
  g1[4] = 512;                                           // tile_dim1 = 512
  g1[5] = 512;                                           // tensor_dim0_stride
  g1[6] = 0;
  g1[7] = 0;
  v4i z4 = {0, 0, 0, 0};
#if defined(__clang_major__) && (__clang_major__ >= 23)
  v8i z8 = {0, 0, 0, 0, 0, 0, 0, 0};
  __builtin_amdgcn_tensor_load_to_lds(g0, g1, z4, z4, z8, 0);
#else
  __builtin_amdgcn_tensor_load_to_lds(g0, g1, z4, z4, 0);
#endif
}
#endif // HAVE_TDM

// ===========================================================================
// Kernel 0: bulk f32 -> bf16 (hardware cvt; 8 elements per thread)
// ===========================================================================
__global__ void __launch_bounds__(256)
cvt_f32_bf16(const float* __restrict__ src, __bf16* __restrict__ dst, int n) {
  int i = (blockIdx.x * 256 + threadIdx.x) * 8;
  if (i + 8 <= n) {
#pragma unroll
    for (int e = 0; e < 8; ++e) dst[i + e] = (__bf16)src[i + e];
  }
}

// ===========================================================================
// Kernel 1: OUT(bf16, BHTD layout) = A(bf16) @ W(bf16).T + b(f32)
// Block = one 16-row M-tile x all 512 N-cols (8 waves x 16x64 strips).
// W K-slabs double-buffered in LDS via TDM; wave0 drives the DMA.
// grid: M_/16 blocks x 256 threads.
// ===========================================================================
__global__ void __launch_bounds__(256)
proj_qkv_kernel(const __bf16* __restrict__ A, const __bf16* __restrict__ W,
                const float* __restrict__ bias, __bf16* __restrict__ Out) {
  __shared__ __bf16 Bs[2][E_ * 32];        // 2 x 32KB W slabs

  const int wave = threadIdx.x >> 5;
  const int lane = threadIdx.x & 31;
  const int tm   = blockIdx.x;             // M-tile
  const int tn64 = wave;                   // N-strip of 64 (8 strips = E_)
  const int m    = lane & 15;
  const int n    = m;
  const int half = lane >> 4;

  const __bf16* arow = A + (size_t)(tm*16 + m) * E_;

#if defined(HAVE_TDM)
  if (wave == 0) tdm_issue_w_slab(W, 0, 0u);
#endif

  v8f acc[4] = {v8f{}, v8f{}, v8f{}, v8f{}};
  for (int ic = 0; ic < NCH; ++ic) {
    const int kb  = ic * 32;
    const int buf = ic & 1;
#if defined(HAVE_TDM)
    if (wave == 0) {
      if (ic + 1 < NCH) {                  // prefetch next slab, wait current
        tdm_issue_w_slab(W, (ic + 1) * 32, (unsigned)(((ic + 1) & 1) * (E_*32*2)));
        __builtin_amdgcn_s_wait_tensorcnt(1);
      } else {
        __builtin_amdgcn_s_wait_tensorcnt(0);
      }
    }
#else
    for (int t = threadIdx.x; t < E_*4; t += 256) {
      const int nrow = t >> 2, c0 = (t & 3) * 8;
#pragma unroll
      for (int e = 0; e < 8; ++e)
        Bs[buf][nrow*32 + c0 + e] = W[(size_t)nrow*E_ + kb + c0 + e];
    }
#endif
    __syncthreads();                       // slab[buf] ready for all waves
    tdm_lds_publish(&Bs[0][0]);            // escape + clobber: keep ds_loads

    v16bf a;
#pragma unroll
    for (int e = 0; e < 8; ++e) {
      a[e]     = arow[kb + 16*half + e];
      a[e + 8] = arow[kb + 16*half + 8 + e];
    }
#pragma unroll
    for (int nb = 0; nb < 4; ++nb) {
      const __bf16* bsrc = &Bs[buf][(tn64*64 + nb*16 + n)*32 + 16*half];
      v16bf b;
#pragma unroll
      for (int e = 0; e < 16; ++e) b[e] = bsrc[e];
      acc[nb] = __builtin_amdgcn_wmma_f32_16x16x32_bf16(false, a, false, b,
                                                        (short)0, acc[nb],
                                                        false, false);
    }
    __syncthreads();                       // done reading slab[buf]
  }

  // store: rows tm*16 + r + 8*half; col = tn64*64 + nb*16 + n = h*64 + d
  const int h = tn64;                      // since D_ == 64
#pragma unroll
  for (int nb = 0; nb < 4; ++nb) {
    const int col = tn64*64 + nb*16 + n;
    const int d   = nb*16 + n;
    const float bv = bias[col];
#pragma unroll
    for (int r = 0; r < 8; ++r) {
      const int row = tm*16 + r + 8*half;  // global m = b*T + t
      const int b_i = row / T_, t = row % T_;
      Out[(((size_t)(b_i*H_ + h))*T_ + t)*D_ + d] = (__bf16)(acc[nb][r] + bv);
    }
  }
}

// ===========================================================================
// Kernel 2: windowed Gaussian attention with per-row top-96 radix select.
// grid = B*H*(T/128); block = 256 (8 waves); one wave = 16 query rows.
// ===========================================================================
__global__ void __launch_bounds__(256)
attn_kernel(const __bf16* __restrict__ Q, const __bf16* __restrict__ K,
            const __bf16* __restrict__ V, const float* __restrict__ log_sigma,
            __bf16* __restrict__ AO) {
  __shared__ __bf16 Pbuf[8][16][PCOLS];    // softmax probs
  __shared__ float  qsq_s[8][16];

  const int wave = threadIdx.x >> 5;
  const int lane = threadIdx.x & 31;
  const int m    = lane & 15;              // A-row / B-col within tile
  const int n    = m;
  const int half = lane >> 4;

  const int qblk = blockIdx.x % (T_/128);
  const int bh   = blockIdx.x / (T_/128);
  const int h    = bh % H_;
  const int q0   = qblk*128 + wave*16;
  const size_t base = (size_t)bh * T_ * D_;

  const float ls    = log_sigma[h];
  const float scale = -0.5f * __expf(-2.0f * ls);   // 1/(-2*sigma^2)

  // zero-pad probability buffer (this wave's region)
  for (int i = lane; i < 16*PCOLS; i += 32)
    Pbuf[wave][i / PCOLS][i % PCOLS] = (__bf16)0.0f;

  // ---- load Q tile (two 16x32 bf16 A-chunks) + ||q||^2 ----
  const __bf16* qrow = Q + base + (size_t)(q0 + m) * D_;
  v16bf qa0, qa1;
  float qp = 0.f;
#pragma unroll
  for (int e = 0; e < 8; ++e) {
    __bf16 u0 = qrow[8*half + e],      u1 = qrow[16 + 8*half + e];
    __bf16 u2 = qrow[32 + 8*half + e], u3 = qrow[48 + 8*half + e];
    qa0[e] = u0; qa0[e+8] = u1;
    qa1[e] = u2; qa1[e+8] = u3;
    float f0 = (float)u0, f1 = (float)u1, f2 = (float)u2, f3 = (float)u3;
    qp += f0*f0 + f1*f1 + f2*f2 + f3*f3;
  }
  qp += __shfl_xor(qp, 16);                // combine complementary K-halves
  if (half == 0) qsq_s[wave][m] = qp;
  __syncthreads();

  // ---- key-tile range for this wave: [max(0,q0-127) .. q0+15] ----
  int lowq = q0 - (WIN_ - 1); if (lowq < 0) lowq = 0;
  const int t0  = lowq >> 4;
  const int nkt = (q0 >> 4) - t0 + 1;      // <= 9
  const unsigned SK_NEGINF = sortkey(-__builtin_inff());

  // ---- scores (sortable-key form): skey[kt][r] for row r+8*half, col kbase+n
  unsigned skey[MAXKT][8];
#pragma unroll
  for (int kt = 0; kt < MAXKT; ++kt) {
    if (kt >= nkt) {
#pragma unroll
      for (int r = 0; r < 8; ++r) skey[kt][r] = SK_NEGINF;
      continue;
    }
    const int kbase = (t0 + kt) * 16;
    const __bf16* krow = K + base + (size_t)(kbase + n) * D_;
    v16bf b0, b1;
    float kp = 0.f;
#pragma unroll
    for (int e = 0; e < 16; ++e) {
      __bf16 u0 = krow[8*half + e + (e & 8)], u1 = krow[32 + 8*half + e + (e & 8)];
      b0[e] = u0; b1[e] = u1;
      float f0 = (float)u0, f1 = (float)u1;
      kp += f0*f0 + f1*f1;
    }
    kp += __shfl_xor(kp, 16);              // full ||k||^2 for column n
    v8f s = {};
    s = __builtin_amdgcn_wmma_f32_16x16x32_bf16(false, qa0, false, b0,
                                                (short)0, s, false, false);
    s = __builtin_amdgcn_wmma_f32_16x16x32_bf16(false, qa1, false, b1,
                                                (short)0, s, false, false);
    const int key = kbase + n;
#pragma unroll
    for (int r = 0; r < 8; ++r) {
      const int qi = q0 + r + 8*half;
      float sc = (qsq_s[wave][r + 8*half] + kp - 2.0f * s[r]) * scale;
      if (key > qi || key < qi - (WIN_ - 1)) sc = -__builtin_inff();
      skey[kt][r] = sortkey(sc);
    }
  }

  // ---- per-row: max, top-96 threshold (radix select), softmax -> LDS ----
#pragma unroll
  for (int r = 0; r < 8; ++r) {
    unsigned mk = 0;
#pragma unroll
    for (int kt = 0; kt < MAXKT; ++kt) mk = (skey[kt][r] > mk) ? skey[kt][r] : mk;
#pragma unroll
    for (int d = 1; d < 16; d <<= 1) {     // reduce within the 16-lane half
      unsigned o = __shfl_xor(mk, d);
      mk = (o > mk) ? o : mk;
    }
    unsigned prefix = 0;                   // kth-largest key via 32-bit radix
    for (int bit = 31; bit >= 0; --bit) {
      const unsigned cand = prefix | (1u << bit);
      int cnt = 0;
#pragma unroll
      for (int kt = 0; kt < MAXKT; ++kt) cnt += (skey[kt][r] >= cand) ? 1 : 0;
      cnt += __shfl_xor(cnt, 1); cnt += __shfl_xor(cnt, 2);
      cnt += __shfl_xor(cnt, 4); cnt += __shfl_xor(cnt, 8);
      if (cnt >= TOPK_) prefix = cand;
    }
    const float mx = keyfloat(mk);
    float pv[MAXKT];
    float ssum = 0.f;
#pragma unroll
    for (int kt = 0; kt < MAXKT; ++kt) {
      float p = 0.f;
      if (skey[kt][r] >= prefix)
        p = __expf(keyfloat(skey[kt][r]) - mx);   // exp(-inf)=0
      pv[kt] = p; ssum += p;
    }
    ssum += __shfl_xor(ssum, 1); ssum += __shfl_xor(ssum, 2);
    ssum += __shfl_xor(ssum, 4); ssum += __shfl_xor(ssum, 8);
    const float inv = (ssum > 0.f) ? 1.0f / ssum : 0.f;
#pragma unroll
    for (int kt = 0; kt < MAXKT; ++kt)
      if (kt < nkt) Pbuf[wave][r + 8*half][kt*16 + n] = (__bf16)(pv[kt] * inv);
  }
  __syncthreads();

  // ---- O = P (16 x <=144) x V (keys x 64), bf16 WMMA, K-chunks of 32 ----
  v8f o[4] = {v8f{}, v8f{}, v8f{}, v8f{}};
  const int nkc = (nkt + 1) >> 1;
  for (int kc = 0; kc < (MAXKT + 1)/2; ++kc) {
    if (kc >= nkc) break;
    const __bf16* prow = &Pbuf[wave][m][kc*32];
    v16bf pa;
#pragma unroll
    for (int e = 0; e < 8; ++e) {
      pa[e]     = prow[8*half + e];
      pa[e + 8] = prow[16 + 8*half + e];
    }
    const int kgb = (t0 + kc*2) * 16;
#pragma unroll
    for (int nb = 0; nb < 4; ++nb) {
      v16bf vb;
#pragma unroll
      for (int e = 0; e < 16; ++e) {
        int key = kgb + 8*half + e + (e & 8);
        if (key >= T_) key = T_ - 1;       // padded cols have P==0
        vb[e] = V[base + (size_t)key * D_ + nb*16 + n];
      }
      o[nb] = __builtin_amdgcn_wmma_f32_16x16x32_bf16(false, pa, false, vb,
                                                      (short)0, o[nb],
                                                      false, false);
    }
  }

  // ---- store attn-out as bf16 in (B,T,E) for the final projection ----
  const int b_i = bh / H_;
#pragma unroll
  for (int nb = 0; nb < 4; ++nb)
#pragma unroll
    for (int r = 0; r < 8; ++r) {
      const int t   = q0 + r + 8*half;
      const int col = h*D_ + nb*16 + n;
      AO[((size_t)b_i*T_ + t)*E_ + col] = (__bf16)o[nb][r];
    }
}

// ===========================================================================
// Kernel 3: OUT(f32) = AO(bf16) @ Wo(bf16).T + bo;  same TDM-staged GEMM
// ===========================================================================
__global__ void __launch_bounds__(256)
proj_out_kernel(const __bf16* __restrict__ A, const __bf16* __restrict__ W,
                const float* __restrict__ bias, float* __restrict__ Out) {
  __shared__ __bf16 Bs[2][E_ * 32];

  const int wave = threadIdx.x >> 5;
  const int lane = threadIdx.x & 31;
  const int tm   = blockIdx.x;
  const int tn64 = wave;
  const int m    = lane & 15;
  const int n    = m;
  const int half = lane >> 4;

  const __bf16* arow = A + (size_t)(tm*16 + m) * E_;

#if defined(HAVE_TDM)
  if (wave == 0) tdm_issue_w_slab(W, 0, 0u);
#endif

  v8f acc[4] = {v8f{}, v8f{}, v8f{}, v8f{}};
  for (int ic = 0; ic < NCH; ++ic) {
    const int kb  = ic * 32;
    const int buf = ic & 1;
#if defined(HAVE_TDM)
    if (wave == 0) {
      if (ic + 1 < NCH) {
        tdm_issue_w_slab(W, (ic + 1) * 32, (unsigned)(((ic + 1) & 1) * (E_*32*2)));
        __builtin_amdgcn_s_wait_tensorcnt(1);
      } else {
        __builtin_amdgcn_s_wait_tensorcnt(0);
      }
    }
#else
    for (int t = threadIdx.x; t < E_*4; t += 256) {
      const int nrow = t >> 2, c0 = (t & 3) * 8;
#pragma unroll
      for (int e = 0; e < 8; ++e)
        Bs[buf][nrow*32 + c0 + e] = W[(size_t)nrow*E_ + kb + c0 + e];
    }
#endif
    __syncthreads();
    tdm_lds_publish(&Bs[0][0]);            // escape + clobber: keep ds_loads

    v16bf a;
#pragma unroll
    for (int e = 0; e < 8; ++e) {
      a[e]     = arow[kb + 16*half + e];
      a[e + 8] = arow[kb + 16*half + 8 + e];
    }
#pragma unroll
    for (int nb = 0; nb < 4; ++nb) {
      const __bf16* bsrc = &Bs[buf][(tn64*64 + nb*16 + n)*32 + 16*half];
      v16bf b;
#pragma unroll
      for (int e = 0; e < 16; ++e) b[e] = bsrc[e];
      acc[nb] = __builtin_amdgcn_wmma_f32_16x16x32_bf16(false, a, false, b,
                                                        (short)0, acc[nb],
                                                        false, false);
    }
    __syncthreads();
  }

#pragma unroll
  for (int nb = 0; nb < 4; ++nb) {
    const int col = tn64*64 + nb*16 + n;
    const float bv = bias[col];
#pragma unroll
    for (int r = 0; r < 8; ++r) {
      const int row = tm*16 + r + 8*half;
      Out[(size_t)row * E_ + col] = acc[nb][r] + bv;
    }
  }
}

// ===========================================================================
extern "C" void kernel_launch(void* const* d_in, const int* in_sizes, int n_in,
                              void* d_out, int out_size, void* d_ws, size_t ws_size,
                              hipStream_t stream) {
  (void)in_sizes; (void)n_in; (void)out_size; (void)ws_size;
  const float* x  = (const float*)d_in[0];
  const float* Wq = (const float*)d_in[1];
  const float* bq = (const float*)d_in[2];
  const float* Wk = (const float*)d_in[3];
  const float* bk = (const float*)d_in[4];
  const float* Wv = (const float*)d_in[5];
  const float* bv = (const float*)d_in[6];
  const float* Wo = (const float*)d_in[7];
  const float* bo = (const float*)d_in[8];
  const float* ls = (const float*)d_in[9];
  float* out = (float*)d_out;

  // workspace (bf16): xb[M*E], wq/wk/wv/wo[E*E], qb/kb/vb[M*E], ao[M*E]
  __bf16* xb  = (__bf16*)d_ws;
  __bf16* wqb = xb  + (size_t)M_ * E_;
  __bf16* wkb = wqb + (size_t)E_ * E_;
  __bf16* wvb = wkb + (size_t)E_ * E_;
  __bf16* wob = wvb + (size_t)E_ * E_;
  __bf16* qb  = wob + (size_t)E_ * E_;
  __bf16* kb  = qb  + (size_t)M_ * E_;
  __bf16* vb  = kb  + (size_t)M_ * E_;
  __bf16* ao  = vb  + (size_t)M_ * E_;

  // 0) one-time conversions
  cvt_f32_bf16<<<(M_*E_)/(8*256), 256, 0, stream>>>(x,  xb,  M_*E_);
  cvt_f32_bf16<<<(E_*E_)/(8*256), 256, 0, stream>>>(Wq, wqb, E_*E_);
  cvt_f32_bf16<<<(E_*E_)/(8*256), 256, 0, stream>>>(Wk, wkb, E_*E_);
  cvt_f32_bf16<<<(E_*E_)/(8*256), 256, 0, stream>>>(Wv, wvb, E_*E_);
  cvt_f32_bf16<<<(E_*E_)/(8*256), 256, 0, stream>>>(Wo, wob, E_*E_);

  // 1) QKV projections (one block per 16-row M-tile; TDM-staged weights)
  proj_qkv_kernel<<<M_/16, 256, 0, stream>>>(xb, wqb, bq, qb);
  proj_qkv_kernel<<<M_/16, 256, 0, stream>>>(xb, wkb, bk, kb);
  proj_qkv_kernel<<<M_/16, 256, 0, stream>>>(xb, wvb, bv, vb);

  // 2) attention
  attn_kernel<<<B_*H_*(T_/128), 256, 0, stream>>>(qb, kb, vb, ls, ao);

  // 3) output projection
  proj_out_kernel<<<M_/16, 256, 0, stream>>>(ao, wob, bo, out);
}
